// FlaxMixtralDecoderLayer_74758200754532
// MI455X (gfx1250) — matrix operand order repaired and checked by reading, not compiled, not run on hardware
//
#include <hip/hip_runtime.h>
#include <hip/hip_bf16.h>
#include <math.h>

typedef __attribute__((ext_vector_type(16))) _Float16 v16h;
typedef __attribute__((ext_vector_type(8)))  float    v8f;
typedef __attribute__((ext_vector_type(4)))  float    f32x4;

#define S_LEN  2048
#define DMODEL 1024
#define NHEAD  16
#define NKV    4
#define HDIM   64
#define FDIM   3584
#define NEXP   8
#define EPSV   1e-5f

// ---------------------------------------------------------------------------
// RMSNorm: fp32 in -> f16 out (WMMA A operand precision)
// ---------------------------------------------------------------------------
__global__ void rmsnorm_f16(const float* __restrict__ x, const float* __restrict__ w,
                            _Float16* __restrict__ o, int D) {
  __shared__ float red[256];
  const int row = blockIdx.x;
  const float* xr = x + (size_t)row * D;
  float ss = 0.f;
  for (int i = threadIdx.x; i < D; i += blockDim.x) { float v = xr[i]; ss += v * v; }
  red[threadIdx.x] = ss;
  __syncthreads();
  for (int st = 128; st > 0; st >>= 1) {
    if ((int)threadIdx.x < st) red[threadIdx.x] += red[threadIdx.x + st];
    __syncthreads();
  }
  const float inv = rsqrtf(red[0] / (float)D + EPSV);
  for (int i = threadIdx.x; i < D; i += blockDim.x)
    o[(size_t)row * D + i] = (_Float16)(xr[i] * inv * w[i]);
}

// ---------------------------------------------------------------------------
// WMMA GEMM: C(MxN,f32) = A(MxK,f16 row-major) * B(KxN,f32 row-major)
// Block = 256 threads (8 waves), block tile 64M x 128N, k-step 32.
// Double-buffered pipeline: async A DMA (global_load_async_to_lds_b128) and
// coalesced fp32 B loads for tile k+32 overlap the 4 WMMAs on tile k;
// one barrier per k-step.
// MODE 0: dense store            MODE 1: dense store + residual
// MODE 2: A rows gathered via token list, C stored compacted (expert up-proj)
// MODE 3: A compacted, C scattered to out[token] * slot weight (down-proj)
// ---------------------------------------------------------------------------
template <int MODE>
__global__ void gemm_a16(const _Float16* __restrict__ A, int lda,
                         const float* __restrict__ B, int ldb,
                         float* __restrict__ C, int ldc, int K,
                         const float* __restrict__ res,
                         const int* __restrict__ agather,
                         const int* __restrict__ cscatter,
                         const float* __restrict__ slotw,
                         const int* __restrict__ cntp) {
  __shared__ _Float16 As[2][64 * 32];    // [m][k] row-major
  __shared__ _Float16 Bs[2][128 * 32];   // [n][k] transposed

  const int tid  = threadIdx.x;
  const int wave = tid >> 5;
  const int lane = tid & 31;
  const int lm = lane & 15, grp = lane >> 4;
  const int wm = wave >> 1;           // 0..3 : 16-row strip
  const int wn = wave & 1;            // 0..1 : 64-col strip
  const int m0 = blockIdx.y * 64;
  const int n0 = blockIdx.x * 128;

  int cnt = 0x7fffffff;
  if (MODE >= 2) {
    cnt = cntp[0];
    if (m0 >= cnt) return;  // uniform: whole block exits before barriers/WMMA
  }

  // Per-thread cooperative-load coordinates
  const int arow = tid >> 2;               // 0..63
  const int acol = (tid & 3) * 8;          // 0,8,16,24
  const unsigned lds_a[2] = {
      (unsigned)(uintptr_t)(&As[0][arow * 32 + acol]),
      (unsigned)(uintptr_t)(&As[1][arow * 32 + acol])};
  const int aslot = m0 + arow;
  const int atok = (MODE == 2) ? agather[aslot] : aslot;  // idx[] zero-filled
  const _Float16* agp = A + (size_t)atok * lda + acol;

  const int brow = tid >> 3;
  (void)brow;

  v8f acc0 = {}, acc1 = {}, acc2 = {}, acc3 = {};

  // ---- prologue: fill buffer 0 for k0 = 0
  asm volatile("global_load_async_to_lds_b128 %0, %1, off"
               :: "v"(lds_a[0]), "v"(agp) : "memory");
#pragma unroll
  for (int j = 0; j < 4; ++j) {
    const int i = tid + j * 256;
    const int br = i >> 5;
    const int bc = (i & 31) * 4;
    const f32x4 f = *(const f32x4*)(B + (size_t)br * ldb + n0 + bc);
    Bs[0][(bc + 0) * 32 + br] = (_Float16)f.x;
    Bs[0][(bc + 1) * 32 + br] = (_Float16)f.y;
    Bs[0][(bc + 2) * 32 + br] = (_Float16)f.z;
    Bs[0][(bc + 3) * 32 + br] = (_Float16)f.w;
  }
  asm volatile("s_wait_asynccnt 0x0" ::: "memory");
  __syncthreads();

  int p = 0;
  for (int k0 = 0; k0 < K; k0 += 32) {
    const bool more = (k0 + 32) < K;   // uniform
    f32x4 nb[4];
    if (more) {
      // issue next A-tile DMA and next B-tile global loads (overlap compute)
      asm volatile("global_load_async_to_lds_b128 %0, %1, off"
                   :: "v"(lds_a[p ^ 1]), "v"(agp + k0 + 32) : "memory");
#pragma unroll
      for (int j = 0; j < 4; ++j) {
        const int i = tid + j * 256;
        const int br = i >> 5;
        const int bc = (i & 31) * 4;
        nb[j] = *(const f32x4*)(B + (size_t)(k0 + 32 + br) * ldb + n0 + bc);
      }
    }

    // ---- compute on buffer p
    v16h a;
#pragma unroll
    for (int j = 0; j < 8; ++j) {
      const int kk = grp * 16 + 2 * j;
      a[2 * j]     = As[p][(wm * 16 + lm) * 32 + kk];
      a[2 * j + 1] = As[p][(wm * 16 + lm) * 32 + kk + 1];
    }
#pragma unroll
    for (int t = 0; t < 4; ++t) {
      const int nn = wn * 64 + t * 16 + lm;
      v16h b;
#pragma unroll
      for (int j = 0; j < 8; ++j) {
        const int kk = grp * 16 + 2 * j;
        b[2 * j]     = Bs[p][nn * 32 + kk];
        b[2 * j + 1] = Bs[p][nn * 32 + kk + 1];
      }
      v8f* ap = (t == 0) ? &acc0 : (t == 1) ? &acc1 : (t == 2) ? &acc2 : &acc3;
      *ap = __builtin_amdgcn_wmma_f32_16x16x32_f16(false, a, false, b, (short)0,
                                                   *ap, false, false);
    }

    if (more) {
      // convert + stage next B tile, drain the async DMA
#pragma unroll
      for (int j = 0; j < 4; ++j) {
        const int i = tid + j * 256;
        const int br = i >> 5;
        const int bc = (i & 31) * 4;
        Bs[p ^ 1][(bc + 0) * 32 + br] = (_Float16)nb[j].x;
        Bs[p ^ 1][(bc + 1) * 32 + br] = (_Float16)nb[j].y;
        Bs[p ^ 1][(bc + 2) * 32 + br] = (_Float16)nb[j].z;
        Bs[p ^ 1][(bc + 3) * 32 + br] = (_Float16)nb[j].w;
      }
      asm volatile("s_wait_asynccnt 0x0" ::: "memory");
    }
    __syncthreads();
    p ^= 1;
  }

#pragma unroll
  for (int t = 0; t < 4; ++t) {
    const v8f* ap = (t == 0) ? &acc0 : (t == 1) ? &acc1 : (t == 2) ? &acc2 : &acc3;
#pragma unroll
    for (int r = 0; r < 8; ++r) {
      const int row = m0 + wm * 16 + r + 8 * grp;
      const int col = n0 + wn * 64 + t * 16 + lm;
      const float val = (*ap)[r];
      if (MODE == 3) {
        if (row < cnt) {
          const int tok = cscatter[row];
          C[(size_t)tok * ldc + col] += slotw[row] * val;
        }
      } else {
        const size_t cidx = (size_t)row * ldc + col;
        if (MODE == 1)
          C[cidx] = val + res[cidx];
        else
          C[cidx] = val;   // MODE 0 dense, MODE 2 compacted
      }
    }
  }
}

// ---------------------------------------------------------------------------
// RoPE + f16 conversion for Q/K, plain conversion for V
// ---------------------------------------------------------------------------
__global__ void rope_cvt(const float* __restrict__ qf, const float* __restrict__ kf,
                         const float* __restrict__ vf, const int* __restrict__ pos,
                         _Float16* __restrict__ q16, _Float16* __restrict__ k16,
                         _Float16* __restrict__ v16) {
  const int s = blockIdx.x;
  const float p = (float)pos[s];
  for (int idx = threadIdx.x; idx < NHEAD * HDIM; idx += blockDim.x) {
    const int d = idx & (HDIM - 1);
    const int i = d & 31;
    const float ang = p * powf(1.0e6f, -(float)(2 * i) / 64.0f);
    const float cs = cosf(ang), sn = sinf(ang);
    const size_t base = (size_t)s * NHEAD * HDIM;
    const float x = qf[base + idx];
    const float other = (d < 32) ? -qf[base + idx + 32] : qf[base + idx - 32];
    q16[base + idx] = (_Float16)(x * cs + other * sn);
  }
  for (int idx = threadIdx.x; idx < NKV * HDIM; idx += blockDim.x) {
    const int d = idx & (HDIM - 1);
    const int i = d & 31;
    const float ang = p * powf(1.0e6f, -(float)(2 * i) / 64.0f);
    const float cs = cosf(ang), sn = sinf(ang);
    const size_t base = (size_t)s * NKV * HDIM;
    const float x = kf[base + idx];
    const float other = (d < 32) ? -kf[base + idx + 32] : kf[base + idx - 32];
    k16[base + idx] = (_Float16)(x * cs + other * sn);
    v16[base + idx] = (_Float16)vf[base + idx];
  }
}

// ---------------------------------------------------------------------------
// Causal flash attention, one wave per (16-row q tile, head).
// ---------------------------------------------------------------------------
__global__ void flash_attn(const _Float16* __restrict__ q, const _Float16* __restrict__ k,
                           const _Float16* __restrict__ v, _Float16* __restrict__ attn) {
  __shared__ _Float16 Pls[16 * 32];
  const int lane = threadIdx.x & 31;
  const int lm = lane & 15, grp = lane >> 4;
  const int m0 = blockIdx.x * 16;
  const int hh = blockIdx.y;
  const int kvh = hh / (NHEAD / NKV);

  v16h aq0, aq1;
  {
    const _Float16* qp = q + ((size_t)(m0 + lm) * NHEAD + hh) * HDIM;
#pragma unroll
    for (int j = 0; j < 8; ++j) {
      const int kk = grp * 16 + 2 * j;
      aq0[2 * j] = qp[kk];      aq0[2 * j + 1] = qp[kk + 1];
      aq1[2 * j] = qp[32 + kk]; aq1[2 * j + 1] = qp[32 + kk + 1];
    }
  }

  float mrow[8], lrow[8];
#pragma unroll
  for (int r = 0; r < 8; ++r) { mrow[r] = -3.0e38f; lrow[r] = 0.f; }
  v8f o0 = {}, o1 = {}, o2 = {}, o3 = {};

  for (int kb = 0; kb < m0 + 16; kb += 32) {
    v8f s0 = {}, s1 = {};
#pragma unroll
    for (int sub = 0; sub < 2; ++sub) {
      const int key = kb + sub * 16 + lm;
      const _Float16* kp = k + ((size_t)key * NKV + kvh) * HDIM;
      v16h b0, b1;
#pragma unroll
      for (int j = 0; j < 8; ++j) {
        const int kk = grp * 16 + 2 * j;
        b0[2 * j] = kp[kk];      b0[2 * j + 1] = kp[kk + 1];
        b1[2 * j] = kp[32 + kk]; b1[2 * j + 1] = kp[32 + kk + 1];
      }
      if (sub == 0) {
        s0 = __builtin_amdgcn_wmma_f32_16x16x32_f16(false, aq0, false, b0, (short)0, s0, false, false);
        s0 = __builtin_amdgcn_wmma_f32_16x16x32_f16(false, aq1, false, b1, (short)0, s0, false, false);
      } else {
        s1 = __builtin_amdgcn_wmma_f32_16x16x32_f16(false, aq0, false, b0, (short)0, s1, false, false);
        s1 = __builtin_amdgcn_wmma_f32_16x16x32_f16(false, aq1, false, b1, (short)0, s1, false, false);
      }
    }

    const float sc = 0.125f;  // 1/sqrt(64)
#pragma unroll
    for (int r = 0; r < 8; ++r) {
      const int row = m0 + r + 8 * grp;
      float x0 = s0[r] * sc; if (kb + lm > row)      x0 = -3.0e38f;
      float x1 = s1[r] * sc; if (kb + 16 + lm > row) x1 = -3.0e38f;
      float bm = fmaxf(x0, x1);
#pragma unroll
      for (int off = 8; off >= 1; off >>= 1) bm = fmaxf(bm, __shfl_xor(bm, off, 16));
      const float nm = fmaxf(mrow[r], bm);
      const float corr = expf(mrow[r] - nm);
      const float p0 = expf(x0 - nm), p1 = expf(x1 - nm);
      float rs = p0 + p1;
#pragma unroll
      for (int off = 8; off >= 1; off >>= 1) rs += __shfl_xor(rs, off, 16);
      lrow[r] = lrow[r] * corr + rs;
      mrow[r] = nm;
      o0[r] *= corr; o1[r] *= corr; o2[r] *= corr; o3[r] *= corr;
      Pls[(r + 8 * grp) * 32 + lm]      = (_Float16)p0;
      Pls[(r + 8 * grp) * 32 + 16 + lm] = (_Float16)p1;
    }
    __syncthreads();

    v16h ap;
#pragma unroll
    for (int j = 0; j < 8; ++j) {
      const int kk = grp * 16 + 2 * j;
      ap[2 * j] = Pls[lm * 32 + kk]; ap[2 * j + 1] = Pls[lm * 32 + kk + 1];
    }
#pragma unroll
    for (int t = 0; t < 4; ++t) {
      v16h bv;
#pragma unroll
      for (int j = 0; j < 8; ++j) {
        const int kk = grp * 16 + 2 * j;
        bv[2 * j]     = v[((size_t)(kb + kk) * NKV + kvh) * HDIM + t * 16 + lm];
        bv[2 * j + 1] = v[((size_t)(kb + kk + 1) * NKV + kvh) * HDIM + t * 16 + lm];
      }
      v8f* op = (t == 0) ? &o0 : (t == 1) ? &o1 : (t == 2) ? &o2 : &o3;
      *op = __builtin_amdgcn_wmma_f32_16x16x32_f16(false, ap, false, bv, (short)0, *op, false, false);
    }
    __syncthreads();
  }

#pragma unroll
  for (int r = 0; r < 8; ++r) {
    const float invl = 1.0f / lrow[r];
    const size_t base = (size_t)(m0 + r + 8 * grp) * (NHEAD * HDIM) + (size_t)hh * HDIM;
    attn[base + 0 * 16 + lm] = (_Float16)(o0[r] * invl);
    attn[base + 1 * 16 + lm] = (_Float16)(o1[r] * invl);
    attn[base + 2 * 16 + lm] = (_Float16)(o2[r] * invl);
    attn[base + 3 * 16 + lm] = (_Float16)(o3[r] * invl);
  }
}

// ---------------------------------------------------------------------------
// Router: logits = h @ w_gate, top-2 softmax; builds per-expert compacted
// token lists (count, token index, routing weight) for gathered MoE GEMMs.
// ---------------------------------------------------------------------------
__global__ void router_topk(const _Float16* __restrict__ h, const float* __restrict__ wg,
                            int* __restrict__ cnt, int* __restrict__ idx,
                            float* __restrict__ ew) {
  const int s = blockIdx.x;
  const int lane = threadIdx.x;
  float acc[NEXP];
#pragma unroll
  for (int e = 0; e < NEXP; ++e) acc[e] = 0.f;
  for (int d = lane; d < DMODEL; d += 32) {
    const float hv = (float)h[(size_t)s * DMODEL + d];
#pragma unroll
    for (int e = 0; e < NEXP; ++e) acc[e] += hv * wg[d * NEXP + e];
  }
#pragma unroll
  for (int e = 0; e < NEXP; ++e)
    for (int off = 16; off >= 1; off >>= 1) acc[e] += __shfl_xor(acc[e], off, 32);
  if (lane == 0) {
    int b1 = 0;
    for (int e = 1; e < NEXP; ++e) if (acc[e] > acc[b1]) b1 = e;
    int b2 = -1;
    for (int e = 0; e < NEXP; ++e) { if (e == b1) continue; if (b2 < 0 || acc[e] > acc[b2]) b2 = e; }
    const float e2 = expf(acc[b2] - acc[b1]);
    const float den = 1.0f + e2;
    const int s1 = atomicAdd(&cnt[b1], 1);
    idx[b1 * S_LEN + s1] = s;
    ew[b1 * S_LEN + s1] = 1.0f / den;
    const int s2 = atomicAdd(&cnt[b2], 1);
    idx[b2 * S_LEN + s2] = s;
    ew[b2 * S_LEN + s2] = e2 / den;
  }
}

__global__ void zero_i32(int* __restrict__ p, int n) {
  const int i = blockIdx.x * blockDim.x + threadIdx.x;
  if (i < n) p[i] = 0;
}

__global__ void copy_f32(const float* __restrict__ src, float* __restrict__ dst, int n) {
  const int i = blockIdx.x * blockDim.x + threadIdx.x;
  if (i < n) dst[i] = src[i];
}

__global__ void silu_mul(const float* __restrict__ g, const float* __restrict__ u,
                         _Float16* __restrict__ t, const int* __restrict__ cntp) {
  const int i = blockIdx.x * blockDim.x + threadIdx.x;
  const int live = cntp[0] * FDIM;   // only routed rows
  if (i < live) {
    const float x = g[i];
    t[i] = (_Float16)((x / (1.0f + expf(-x))) * u[i]);
  }
}

// ---------------------------------------------------------------------------
extern "C" void kernel_launch(void* const* d_in, const int* in_sizes, int n_in,
                              void* d_out, int out_size, void* d_ws, size_t ws_size,
                              hipStream_t stream) {
  (void)in_sizes; (void)n_in; (void)out_size; (void)ws_size;
  const float* hs  = (const float*)d_in[0];
  const int*   pos = (const int*)d_in[2];
  const float* wq  = (const float*)d_in[3];
  const float* wk  = (const float*)d_in[4];
  const float* wv  = (const float*)d_in[5];
  const float* wo  = (const float*)d_in[6];
  const float* wg  = (const float*)d_in[7];
  const float* w1  = (const float*)d_in[8];
  const float* w3  = (const float*)d_in[9];
  const float* w2  = (const float*)d_in[10];
  const float* ln1 = (const float*)d_in[11];
  const float* ln2 = (const float*)d_in[12];
  float* out = (float*)d_out;

  char* base = (char*)d_ws;
  size_t off = 0;
  auto alloc = [&](size_t bytes) -> void* {
    void* r = base + off;
    off = (off + bytes + 255) & ~(size_t)255;
    return r;
  };
  _Float16* h1_16  = (_Float16*)alloc((size_t)S_LEN * DMODEL * 2);
  float*    qf     = (float*)   alloc((size_t)S_LEN * NHEAD * HDIM * 4);
  float*    kf     = (float*)   alloc((size_t)S_LEN * NKV * HDIM * 4);
  float*    vf     = (float*)   alloc((size_t)S_LEN * NKV * HDIM * 4);
  _Float16* q16    = (_Float16*)alloc((size_t)S_LEN * NHEAD * HDIM * 2);
  _Float16* k16    = (_Float16*)alloc((size_t)S_LEN * NKV * HDIM * 2);
  _Float16* v16    = (_Float16*)alloc((size_t)S_LEN * NKV * HDIM * 2);
  _Float16* attn16 = (_Float16*)alloc((size_t)S_LEN * NHEAD * HDIM * 2);
  float*    hid2   = (float*)   alloc((size_t)S_LEN * DMODEL * 4);
  _Float16* h2_16  = (_Float16*)alloc((size_t)S_LEN * DMODEL * 2);
  int*      ecnt   = (int*)     alloc((size_t)NEXP * 4);
  int*      eidx   = (int*)     alloc((size_t)NEXP * S_LEN * 4);
  float*    ewgt   = (float*)   alloc((size_t)NEXP * S_LEN * 4);
  float*    gbuf   = (float*)   alloc((size_t)S_LEN * FDIM * 4);
  float*    ubuf   = (float*)   alloc((size_t)S_LEN * FDIM * 4);
  _Float16* t16    = (_Float16*)alloc((size_t)S_LEN * FDIM * 2);

  // ---- attention block
  rmsnorm_f16<<<S_LEN, 256, 0, stream>>>(hs, ln1, h1_16, DMODEL);
  gemm_a16<0><<<dim3((NHEAD * HDIM) / 128, S_LEN / 64), 256, 0, stream>>>(
      h1_16, DMODEL, wq, NHEAD * HDIM, qf, NHEAD * HDIM, DMODEL,
      nullptr, nullptr, nullptr, nullptr, nullptr);
  gemm_a16<0><<<dim3((NKV * HDIM) / 128, S_LEN / 64), 256, 0, stream>>>(
      h1_16, DMODEL, wk, NKV * HDIM, kf, NKV * HDIM, DMODEL,
      nullptr, nullptr, nullptr, nullptr, nullptr);
  gemm_a16<0><<<dim3((NKV * HDIM) / 128, S_LEN / 64), 256, 0, stream>>>(
      h1_16, DMODEL, wv, NKV * HDIM, vf, NKV * HDIM, DMODEL,
      nullptr, nullptr, nullptr, nullptr, nullptr);
  rope_cvt<<<S_LEN, 256, 0, stream>>>(qf, kf, vf, pos, q16, k16, v16);
  flash_attn<<<dim3(S_LEN / 16, NHEAD), 32, 0, stream>>>(q16, k16, v16, attn16);
  gemm_a16<1><<<dim3(DMODEL / 128, S_LEN / 64), 256, 0, stream>>>(
      attn16, NHEAD * HDIM, wo, DMODEL, hid2, DMODEL, NHEAD * HDIM,
      hs, nullptr, nullptr, nullptr, nullptr);

  // ---- MoE block
  rmsnorm_f16<<<S_LEN, 256, 0, stream>>>(hid2, ln2, h2_16, DMODEL);
  zero_i32<<<(NEXP + 255) / 256, 256, 0, stream>>>(ecnt, NEXP);
  zero_i32<<<(NEXP * S_LEN + 255) / 256, 256, 0, stream>>>(eidx, NEXP * S_LEN);
  router_topk<<<S_LEN, 32, 0, stream>>>(h2_16, wg, ecnt, eidx, ewgt);
  copy_f32<<<(S_LEN * DMODEL + 255) / 256, 256, 0, stream>>>(hid2, out, S_LEN * DMODEL);

  for (int e = 0; e < NEXP; ++e) {
    // up-projections on gathered tokens only (top-2 routing: ~S/4 rows/expert)
    gemm_a16<2><<<dim3(FDIM / 128, S_LEN / 64), 256, 0, stream>>>(
        h2_16, DMODEL, w1 + (size_t)e * DMODEL * FDIM, FDIM, gbuf, FDIM, DMODEL,
        nullptr, eidx + (size_t)e * S_LEN, nullptr, nullptr, ecnt + e);
    gemm_a16<2><<<dim3(FDIM / 128, S_LEN / 64), 256, 0, stream>>>(
        h2_16, DMODEL, w3 + (size_t)e * DMODEL * FDIM, FDIM, ubuf, FDIM, DMODEL,
        nullptr, eidx + (size_t)e * S_LEN, nullptr, nullptr, ecnt + e);
    silu_mul<<<(S_LEN * FDIM + 255) / 256, 256, 0, stream>>>(gbuf, ubuf, t16, ecnt + e);
    // down-projection, scatter-accumulate weighted rows into out[token]
    gemm_a16<3><<<dim3(DMODEL / 128, S_LEN / 64), 256, 0, stream>>>(
        t16, FDIM, w2 + (size_t)e * FDIM * DMODEL, DMODEL, out, DMODEL, FDIM,
        nullptr, nullptr, eidx + (size_t)e * S_LEN, ewgt + (size_t)e * S_LEN, ecnt + e);
  }
}
